// SlotAttention_15908558865301
// MI455X (gfx1250) — matrix operand order repaired and checked by reading, not compiled, not run on hardware
//
#include <hip/hip_runtime.h>
#include <math.h>

// ---------------------------------------------------------------------------
// Slot Attention for MI455X (gfx1250).
// B=64, N=4096, D=128, S=8, H=256, 3 iterations.
// Memory-bound: keep k/v in bf16 (128 MiB total -> resident in 192MB L2),
// big GEMMs via v_wmma_f32_16x16x32_bf16 (f32 accumulate), tiny slot math f32.
// attn_update_kernel is defined FIRST so the disasm window shows its k-tile
// staging path (TDM tensor_load_to_lds vs fallback copies).
// ---------------------------------------------------------------------------

#define BB 64
#define NN 4096
#define DD 128
#define SS 8
#define HH 256
#define NTILE 64

typedef __attribute__((ext_vector_type(16))) __bf16 v16bf;
typedef __attribute__((ext_vector_type(8)))  __bf16 v8bf;
typedef __attribute__((ext_vector_type(8)))  float  v8f;
typedef __attribute__((ext_vector_type(4)))  unsigned int v4u;
typedef __attribute__((ext_vector_type(8)))  int v8i;
typedef __attribute__((ext_vector_type(4)))  int v4i;

__device__ __forceinline__ __bf16 f2bf(float f) { return (__bf16)f; }

// A fragment (16x32 bf16) from row-major bf16 storage (row_stride in elems).
__device__ __forceinline__ v16bf load_a_row(const __bf16* base, int row_stride, int lane) {
  int m = lane & 15;
  int koff = (lane >> 4) << 3; // 0 or 8
  const __bf16* p = base + m * row_stride + koff;
  v8bf c0 = *(const v8bf*)(p);
  v8bf c1 = *(const v8bf*)(p + 16);
  return __builtin_shufflevector(c0, c1, 0,1,2,3,4,5,6,7,8,9,10,11,12,13,14,15);
}

// B fragment (32x16 bf16) from column-major storage (col_stride in elems).
__device__ __forceinline__ v16bf load_b_col(const __bf16* base, int col_stride, int lane) {
  int n = lane & 15;
  int kb = (lane >> 4) << 3; // 0 or 8
  const __bf16* p = base + n * col_stride + kb;
  v8bf c0 = *(const v8bf*)(p);
  v8bf c1 = *(const v8bf*)(p + 16);
  return __builtin_shufflevector(c0, c1, 0,1,2,3,4,5,6,7,8,9,10,11,12,13,14,15);
}

// ---------------------------------------------------------------------------
// Attention pass: one block per (b, 64-row n-tile). 256 threads = 8 waves.
// k tile staged via Tensor Data Mover when available (issue -> overlap with
// v-transpose staging -> s_wait_tensorcnt -> barrier); logits = k @ q^T via
// WMMA, softmax over 8 slots, then
//   attn_sum[b][s]  += sum_n attn[n][s]           (global f32 atomics)
//   U[b][s][d]      += attn^T @ v  tile-partial   (WMMA + global f32 atomics)
// ---------------------------------------------------------------------------
__global__ void __launch_bounds__(256) attn_update_kernel(
    const __bf16* __restrict__ k_g, const __bf16* __restrict__ v_g,
    const __bf16* __restrict__ q_g, float* __restrict__ U,
    float* __restrict__ attn_sum) {
  __shared__ __bf16 ks[NTILE][DD];   // 16 KB row-major k tile
  __shared__ __bf16 vT[DD][NTILE];   // 16 KB transposed v tile (col-major B)
  __shared__ float lg[NTILE][16];    // logits
  __shared__ __bf16 aT[16][NTILE];   // attn^T (A operand), rows 8..15 zero
  __shared__ float ssum[16];

  const int t = threadIdx.x;
  const int b = blockIdx.y;
  const int n0 = blockIdx.x * NTILE;
  const size_t base = ((size_t)b * NN + n0) * DD;

  // keep the tiny q block warm in near caches (global_prefetch_b8)
  __builtin_prefetch(q_g + (size_t)b * 16 * DD, 0, 1);

#if __has_builtin(__builtin_amdgcn_tensor_load_to_lds)
  const bool tdm_wave = (t < 32);
  // ---- issue TDM for the contiguous 16 KB k tile from wave 0 ----
  if (tdm_wave) {
    const unsigned long long ga = (unsigned long long)(const void*)(k_g + base);
    // LDS aperture truncates flat addresses to addr[31:0] == LDS byte offset
    const unsigned lds_off = (unsigned)(unsigned long long)(uintptr_t)&ks[0][0];
    // D# group 0: count=1 | lds_addr | global_addr[56:0] | type=2
    v4u g0 = {1u, lds_off, (unsigned)(ga & 0xffffffffu),
              ((unsigned)((ga >> 32) & 0x01ffffffu)) | 0x80000000u};
    // D# group 1: 1-D tile of 8192 bf16 elements (data_size=2B)
    const unsigned ne = NTILE * DD; // 8192 elements
    v8i g1 = {(int)(1u << 16),        // data_size=1 (2 bytes)
              (int)(ne << 16),        // tensor_dim0[15:0] in [31:16]
              (int)(1u << 16),        // tensor_dim0 hi=0, tensor_dim1=1
              (int)(ne << 16),        // tensor_dim1 hi=0, tile_dim0=8192
              0,                      // tile_dim1=0 (unused), tile_dim2=0
              (int)ne,                // tensor_dim0_stride lo
              0, 0};                  // stride hi, tensor_dim1_stride
    v4i g2 = {0, 0, 0, 0};
    v4i g3 = {0, 0, 0, 0};
#if defined(__clang_major__) && (__clang_major__ >= 23)
    v8i g4 = {0, 0, 0, 0, 0, 0, 0, 0};
    __builtin_amdgcn_tensor_load_to_lds(g0, g1, g2, g3, g4, 0);
#else
    __builtin_amdgcn_tensor_load_to_lds(g0, g1, g2, g3, 0);
#endif
  }
#else
  // fallback: coalesced dword copies
  {
    const uint32_t* src = (const uint32_t*)(k_g + base);
    uint32_t* dst = (uint32_t*)&ks[0][0];
    for (int i = t; i < NTILE * DD / 2; i += 256) dst[i] = src[i];
  }
#endif

  // stage v tile transposed (all 8 waves; overlaps with the k-tile DMA)
  {
    const int row = t >> 2, q = t & 3;
    const __bf16* vp = v_g + base + (size_t)row * DD + q * 32;
    v8bf tmp[4];
#pragma unroll
    for (int c = 0; c < 4; ++c) tmp[c] = *(const v8bf*)(vp + c * 8);
#pragma unroll
    for (int c = 0; c < 4; ++c)
#pragma unroll
      for (int i = 0; i < 8; ++i) vT[q * 32 + c * 8 + i][row] = tmp[c][i];
  }
  if (t < 16) ssum[t] = 0.0f;

#if __has_builtin(__builtin_amdgcn_tensor_load_to_lds)
  if (tdm_wave) __builtin_amdgcn_s_wait_tensorcnt(0);
#endif
  __syncthreads();

  const int w = t >> 5, lane = t & 31;

  // ---- logits: waves 0..3 each produce a 16(n) x 16(s) tile ----
  if (w < 4) {
    const int rb = w;
    v8f acc = {};
    const __bf16* qb = q_g + (size_t)b * 16 * DD;
#pragma unroll
    for (int kb = 0; kb < 4; ++kb) {
      v16bf a = load_a_row(&ks[rb * 16][0] + kb * 32, DD, lane);
      v16bf bf = load_b_col(qb + kb * 32, DD, lane);
      acc = __builtin_amdgcn_wmma_f32_16x16x32_bf16(false, a, false, bf,
                                                    (short)0, acc, false, false);
    }
    const float scale = 0.08838834764831845f; // 1/sqrt(128)
    const int col = lane & 15;
    const int rbase = rb * 16 + ((lane < 16) ? 0 : 8);
#pragma unroll
    for (int i = 0; i < 8; ++i) lg[rbase + i][col] = acc[i] * scale;
  }
  __syncthreads();

  // ---- softmax over slots (8 valid), build attn^T, per-slot column sums ----
  if (t < NTILE) {
    float mx = -1e30f;
#pragma unroll
    for (int s = 0; s < SS; ++s) mx = fmaxf(mx, lg[t][s]);
    float e[SS], sum = 0.f;
#pragma unroll
    for (int s = 0; s < SS; ++s) { e[s] = __expf(lg[t][s] - mx); sum += e[s]; }
    const float inv = 1.0f / sum;
#pragma unroll
    for (int s = 0; s < SS; ++s) {
      const float a = e[s] * inv;
      aT[s][t] = f2bf(a);
      atomicAdd(&ssum[s], a);
    }
#pragma unroll
    for (int s = SS; s < 16; ++s) aT[s][t] = f2bf(0.0f);
  }
  __syncthreads();
  if (t < 16) atomicAdd(&attn_sum[b * 16 + t], ssum[t]);

  // ---- U partial: all 8 waves, wave w -> 16 d-columns; K = 64 n (2 steps) --
  {
    const int cb = w;
    v8f acc = {};
#pragma unroll
    for (int kb = 0; kb < 2; ++kb) {
      v16bf a = load_a_row(&aT[0][0] + kb * 32, NTILE, lane);
      v16bf bf = load_b_col(&vT[cb * 16][0] + kb * 32, NTILE, lane);
      acc = __builtin_amdgcn_wmma_f32_16x16x32_bf16(false, a, false, bf,
                                                    (short)0, acc, false, false);
    }
    const int col = cb * 16 + (lane & 15);
    const int rbase = (lane < 16) ? 0 : 8;
#pragma unroll
    for (int i = 0; i < 8; ++i)
      atomicAdd(&U[((size_t)b * 16 + rbase + i) * DD + col], acc[i]);
  }
}

// ---------------------------------------------------------------------------
__global__ void cvt_w_kernel(const float* __restrict__ wk, const float* __restrict__ wv,
                             __bf16* __restrict__ wk_bf, __bf16* __restrict__ wv_bf) {
  int i = blockIdx.x * blockDim.x + threadIdx.x;
  if (i < DD * DD) {
    wk_bf[i] = f2bf(wk[i]);
    wv_bf[i] = f2bf(wv[i]);
  }
}

__global__ void init_slots_kernel(const float* __restrict__ noise, const float* __restrict__ mu,
                                  const float* __restrict__ lsig, float* __restrict__ slots) {
  int i = blockIdx.x * blockDim.x + threadIdx.x; // [0, B*S*D)
  if (i < BB * SS * DD) {
    int rem = i % (SS * DD);
    slots[i] = mu[rem] + __expf(lsig[rem]) * noise[rem];
  }
}

__global__ void zero_kernel(float* __restrict__ U, float* __restrict__ attn_sum) {
  int i = blockIdx.x * blockDim.x + threadIdx.x;
  int nu = BB * 16 * DD;
  if (i < nu) U[i] = 0.0f;
  else if (i < nu + BB * 16) attn_sum[i - nu] = 0.0f;
}

// ---------------------------------------------------------------------------
// Fused LayerNorm + K/V projection.  One block per 64 rows of [B*N, D].
// ---------------------------------------------------------------------------
__global__ void __launch_bounds__(256) ln_kv_kernel(
    const float* __restrict__ inputs, const float* __restrict__ ni_g,
    const float* __restrict__ ni_b, const __bf16* __restrict__ wk_bf,
    const float* __restrict__ bk, const __bf16* __restrict__ wv_bf,
    const float* __restrict__ bv, __bf16* __restrict__ k_out,
    __bf16* __restrict__ v_out) {
  __shared__ __bf16 xs[NTILE][DD];   // 16 KB LN'ed inputs (bf16)
  __shared__ __bf16 wls[DD][DD];     // 32 KB one weight matrix at a time

  const int t = threadIdx.x;
  const int row0 = blockIdx.x * NTILE;

  {
    const int row = t >> 2, q = t & 3;
    const float* ip = inputs + (size_t)(row0 + row) * DD + q * 32;
    float v[32];
    float sum = 0.f, ssq = 0.f;
#pragma unroll
    for (int i = 0; i < 32; ++i) { v[i] = ip[i]; sum += v[i]; ssq += v[i] * v[i]; }
    sum += __shfl_xor(sum, 1); sum += __shfl_xor(sum, 2);
    ssq += __shfl_xor(ssq, 1); ssq += __shfl_xor(ssq, 2);
    const float mean = sum * (1.0f / DD);
    const float var = ssq * (1.0f / DD) - mean * mean;
    const float rstd = rsqrtf(var + 1e-5f);
#pragma unroll
    for (int i = 0; i < 32; ++i) {
      int d = q * 32 + i;
      xs[row][d] = f2bf((v[i] - mean) * rstd * ni_g[d] + ni_b[d]);
    }
  }
  __syncthreads();

  const int w = t >> 5, lane = t & 31;
  const int rb = w & 3;
  const int cb0 = (w >> 2) * 4;

  const __bf16* Wm[2] = {wk_bf, wv_bf};
  const float* Bm[2] = {bk, bv};
  __bf16* Om[2] = {k_out, v_out};

  for (int mm = 0; mm < 2; ++mm) {
    {
      const uint32_t* src = (const uint32_t*)Wm[mm];
      uint32_t* dst = (uint32_t*)&wls[0][0];
      for (int i = t; i < DD * DD / 2; i += 256) dst[i] = src[i];
    }
    __syncthreads();

    v16bf afrag[4];
#pragma unroll
    for (int kb = 0; kb < 4; ++kb)
      afrag[kb] = load_a_row(&xs[rb * 16][0] + kb * 32, DD, lane);

#pragma unroll
    for (int j = 0; j < 4; ++j) {
      const int cb = cb0 + j;
      v8f acc = {};
#pragma unroll
      for (int kb = 0; kb < 4; ++kb) {
        v16bf bfrag = load_b_col(&wls[cb * 16][0] + kb * 32, DD, lane);
        acc = __builtin_amdgcn_wmma_f32_16x16x32_bf16(false, afrag[kb], false, bfrag,
                                                      (short)0, acc, false, false);
      }
      const int col = cb * 16 + (lane & 15);
      const int rbase = rb * 16 + ((lane < 16) ? 0 : 8);
      const float bias = Bm[mm][col];
      __bf16* op = Om[mm];
#pragma unroll
      for (int i = 0; i < 8; ++i)
        op[(size_t)(row0 + rbase + i) * DD + col] = f2bf(acc[i] + bias);
    }
    __syncthreads();
  }
}

// ---------------------------------------------------------------------------
// q = LN(slots) @ wq.T + bq  ->  bf16 [B][16][128] (rows 8..15 zero-padded)
// ---------------------------------------------------------------------------
__global__ void __launch_bounds__(256) q_kernel(
    const float* __restrict__ slots, const float* __restrict__ ns_g,
    const float* __restrict__ ns_b, const float* __restrict__ wq,
    const float* __restrict__ bq, __bf16* __restrict__ q_g) {
  __shared__ float sl[SS][DD];
  const int b = blockIdx.x, t = threadIdx.x;

  if (t < 32) {
    const int row = t >> 2, q = t & 3;
    const float* ip = slots + ((size_t)b * SS + row) * DD + q * 32;
    float v[32];
    float sum = 0.f, ssq = 0.f;
#pragma unroll
    for (int i = 0; i < 32; ++i) { v[i] = ip[i]; sum += v[i]; ssq += v[i] * v[i]; }
    sum += __shfl_xor(sum, 1); sum += __shfl_xor(sum, 2);
    ssq += __shfl_xor(ssq, 1); ssq += __shfl_xor(ssq, 2);
    const float mean = sum * (1.0f / DD);
    const float rstd = rsqrtf(ssq * (1.0f / DD) - mean * mean + 1e-5f);
#pragma unroll
    for (int i = 0; i < 32; ++i) {
      int d = q * 32 + i;
      sl[row][d] = (v[i] - mean) * rstd * ns_g[d] + ns_b[d];
    }
  }
  __syncthreads();

  for (int o = t; o < SS * DD; o += 256) {
    const int s = o >> 7, d = o & 127;
    const float* wr = wq + d * DD;
    float acc = bq[d];
#pragma unroll 4
    for (int j = 0; j < DD; ++j) acc += sl[s][j] * wr[j];
    q_g[((size_t)b * 16 + s) * DD + d] = f2bf(acc);
  }
  for (int o = t; o < 8 * DD; o += 256)
    q_g[((size_t)b * 16 + 8) * DD + o] = f2bf(0.0f);
}

// ---------------------------------------------------------------------------
// Slot-side update: normalize updates, GRU cell, LN + GELU MLP residual.
// ---------------------------------------------------------------------------
__global__ void __launch_bounds__(256) slot_update_kernel(
    float* __restrict__ slots, const float* __restrict__ U,
    const float* __restrict__ attn_sum, const float* __restrict__ wih,
    const float* __restrict__ whh, const float* __restrict__ bih,
    const float* __restrict__ bhh, const float* __restrict__ ln_g,
    const float* __restrict__ ln_b, const float* __restrict__ w1,
    const float* __restrict__ b1, const float* __restrict__ w2,
    const float* __restrict__ b2) {
  __shared__ float upd[SS][DD];
  __shared__ float sl[SS][DD];
  __shared__ float gi[SS][3 * DD];
  __shared__ float gh[SS][3 * DD];
  __shared__ float nsl[SS][DD];
  __shared__ float hh[SS][HH];
  __shared__ float mu_s[SS], rs_s[SS];

  const int b = blockIdx.x, t = threadIdx.x;

  __builtin_prefetch(wih, 0, 1);
  __builtin_prefetch(whh, 0, 1);

  for (int o = t; o < SS * DD; o += 256) {
    const int s = o >> 7, d = o & 127;
    float dnm = fmaxf(attn_sum[b * 16 + s], 1e-8f);
    upd[s][d] = U[((size_t)b * 16 + s) * DD + d] / dnm;
    sl[s][d] = slots[((size_t)b * SS + s) * DD + d];
  }
  __syncthreads();

  for (int o = t; o < SS * 3 * DD; o += 256) {
    const int s = o / (3 * DD), j = o % (3 * DD);
    const float* wi = wih + (size_t)j * DD;
    const float* wh = whh + (size_t)j * DD;
    float a = bih[j], c = bhh[j];
#pragma unroll 4
    for (int d = 0; d < DD; ++d) { a += upd[s][d] * wi[d]; c += sl[s][d] * wh[d]; }
    gi[s][j] = a; gh[s][j] = c;
  }
  __syncthreads();

  for (int o = t; o < SS * DD; o += 256) {
    const int s = o >> 7, d = o & 127;
    const float r = 1.0f / (1.0f + __expf(-(gi[s][d] + gh[s][d])));
    const float z = 1.0f / (1.0f + __expf(-(gi[s][DD + d] + gh[s][DD + d])));
    const float nn = tanhf(gi[s][2 * DD + d] + r * gh[s][2 * DD + d]);
    nsl[s][d] = (1.0f - z) * nn + z * sl[s][d];
  }
  __syncthreads();

  if (t < SS) {
    float sm = 0.f, ss2 = 0.f;
    for (int d = 0; d < DD; ++d) { float x = nsl[t][d]; sm += x; ss2 += x * x; }
    const float m = sm * (1.0f / DD);
    mu_s[t] = m;
    rs_s[t] = rsqrtf(ss2 * (1.0f / DD) - m * m + 1e-5f);
  }
  __syncthreads();
  for (int o = t; o < SS * DD; o += 256) {
    const int s = o >> 7, d = o & 127;
    upd[s][d] = (nsl[s][d] - mu_s[s]) * rs_s[s] * ln_g[d] + ln_b[d];
  }
  __syncthreads();

  for (int o = t; o < SS * HH; o += 256) {
    const int s = o >> 8, j = o & 255;
    const float* wr = w1 + (size_t)j * DD;
    float a = b1[j];
#pragma unroll 4
    for (int d = 0; d < DD; ++d) a += upd[s][d] * wr[d];
    hh[s][j] = 0.5f * a * (1.0f + erff(a * 0.7071067811865475f));
  }
  __syncthreads();

  for (int o = t; o < SS * DD; o += 256) {
    const int s = o >> 7, d = o & 127;
    const float* wr = w2 + (size_t)d * HH;
    float a = b2[d];
#pragma unroll 4
    for (int j = 0; j < HH; ++j) a += hh[s][j] * wr[j];
    slots[((size_t)b * SS + s) * DD + d] = nsl[s][d] + a;
  }
}

// ---------------------------------------------------------------------------
extern "C" void kernel_launch(void* const* d_in, const int* in_sizes, int n_in,
                              void* d_out, int out_size, void* d_ws, size_t ws_size,
                              hipStream_t stream) {
  const float* inputs   = (const float*)d_in[0];
  const float* noise    = (const float*)d_in[1];
  const float* mu       = (const float*)d_in[2];
  const float* lsig     = (const float*)d_in[3];
  const float* wq       = (const float*)d_in[4];
  const float* bq       = (const float*)d_in[5];
  const float* wk       = (const float*)d_in[6];
  const float* bk       = (const float*)d_in[7];
  const float* wv       = (const float*)d_in[8];
  const float* bv       = (const float*)d_in[9];
  const float* gru_wih  = (const float*)d_in[10];
  const float* gru_whh  = (const float*)d_in[11];
  const float* gru_bih  = (const float*)d_in[12];
  const float* gru_bhh  = (const float*)d_in[13];
  const float* ns_g     = (const float*)d_in[14];
  const float* ns_b     = (const float*)d_in[15];
  const float* ni_g     = (const float*)d_in[16];
  const float* ni_b     = (const float*)d_in[17];
  const float* mlp_ln_g = (const float*)d_in[18];
  const float* mlp_ln_b = (const float*)d_in[19];
  const float* mlp_w1   = (const float*)d_in[20];
  const float* mlp_b1   = (const float*)d_in[21];
  const float* mlp_w2   = (const float*)d_in[22];
  const float* mlp_b2   = (const float*)d_in[23];

  char* ws = (char*)d_ws;
  const size_t OFF_K  = 0;                         // 64 MiB bf16 k
  const size_t OFF_V  = 64ull << 20;               // 64 MiB bf16 v
  const size_t OFF_WK = 128ull << 20;              // 32 KB
  const size_t OFF_WV = OFF_WK + 32768;            // 32 KB
  const size_t OFF_Q  = OFF_WK + 65536;            // 256 KB bf16 q
  const size_t OFF_SL = OFF_Q + 262144;            // 256 KB f32 slots
  const size_t OFF_U  = OFF_SL + 262144;           // 512 KB f32 U
  const size_t OFF_AS = OFF_U + 524288;            // 4 KB f32 attn_sum

  __bf16* k_bf  = (__bf16*)(ws + OFF_K);
  __bf16* v_bf  = (__bf16*)(ws + OFF_V);
  __bf16* wk_bf = (__bf16*)(ws + OFF_WK);
  __bf16* wv_bf = (__bf16*)(ws + OFF_WV);
  __bf16* q_bf  = (__bf16*)(ws + OFF_Q);
  float* slots  = (float*)(ws + OFF_SL);
  float* U      = (float*)(ws + OFF_U);
  float* asum   = (float*)(ws + OFF_AS);

  cvt_w_kernel<<<(DD * DD + 255) / 256, 256, 0, stream>>>(wk, wv, wk_bf, wv_bf);
  init_slots_kernel<<<(BB * SS * DD + 255) / 256, 256, 0, stream>>>(noise, mu, lsig, slots);
  ln_kv_kernel<<<(BB * NN) / NTILE, 256, 0, stream>>>(inputs, ni_g, ni_b, wk_bf, bk,
                                                      wv_bf, bv, k_bf, v_bf);
  for (int it = 0; it < 3; ++it) {
    q_kernel<<<BB, 256, 0, stream>>>(slots, ns_g, ns_b, wq, bq, q_bf);
    zero_kernel<<<(BB * 16 * DD + BB * 16 + 255) / 256, 256, 0, stream>>>(U, asum);
    attn_update_kernel<<<dim3(NN / NTILE, BB), 256, 0, stream>>>(k_bf, v_bf, q_bf, U, asum);
    slot_update_kernel<<<BB, 256, 0, stream>>>(slots, U, asum, gru_wih, gru_whh, gru_bih,
                                               gru_bhh, mlp_ln_g, mlp_ln_b, mlp_w1, mlp_b1,
                                               mlp_w2, mlp_b2);
  }
  hipMemcpyAsync(d_out, slots, (size_t)BB * SS * DD * sizeof(float),
                 hipMemcpyDeviceToDevice, stream);
}
